// MoELayer_50740743635377
// MI455X (gfx1250) — matrix operand order repaired and checked by reading, not compiled, not run on hardware
//
#include <hip/hip_runtime.h>
#include <hip/hip_bf16.h>

// ---------------------------------------------------------------------------
// MoE layer for MI455X (gfx1250): sparse top-2 routing + bf16 WMMA GEMMs.
//   N=8192 tokens, D=2048, E=8 experts, H=2048, O=2048, k=2
// ---------------------------------------------------------------------------

#define NTOK 8192
#define DIM  2048
#define NEXP 8
#define HID  2048
#define OUTD 2048

#define BM 128     // token tile
#define BN 128     // output-col tile
#define BK 32      // K step == WMMA K for bf16
#define APAD 8     // LDS row pad -> row stride 40*2 = 80B (8/16B aligned)
#define CPAD 8     // C-tile pad  -> row stride 136*2 = 272B (16B aligned)

typedef __attribute__((ext_vector_type(16))) __bf16 v16bf;
typedef __attribute__((ext_vector_type(8)))  __bf16 v8bf;
typedef __attribute__((ext_vector_type(4)))  __bf16 v4bf;
typedef __attribute__((ext_vector_type(8)))  float  v8f;
typedef __attribute__((ext_vector_type(4)))  float  v4f;
typedef __attribute__((ext_vector_type(4)))  int    v4i;

#define GAS1 __attribute__((address_space(1)))
#define GAS3 __attribute__((address_space(3)))

#ifndef __has_builtin
#define __has_builtin(x) 0
#endif
#if __has_builtin(__builtin_amdgcn_global_load_async_to_lds_b128)
#define HAS_ASYNC_LDS 1
#else
#define HAS_ASYNC_LDS 0
#endif

static __device__ __forceinline__ v16bf comb16(v8bf lo, v8bf hi) {
    v16bf r;
#pragma unroll
    for (int i = 0; i < 8; ++i) { r[i] = lo[i]; r[i + 8] = hi[i]; }
    return r;
}

static __device__ __forceinline__ v4bf cvt4(v4f v) {
    return __builtin_convertvector(v, v4bf);   // -> 2x v_cvt_pk_bf16_f32
}

static __device__ __forceinline__ void wait_async0() {
#if HAS_ASYNC_LDS
#if __has_builtin(__builtin_amdgcn_s_wait_asynccnt)
    __builtin_amdgcn_s_wait_asynccnt(0);
#else
    asm volatile("s_wait_asynccnt 0" ::: "memory");
#endif
#endif
}

// ---------------------------------------------------------------------------
// Kernel 0: zero output (atomic target) + expert counters
// ---------------------------------------------------------------------------
__global__ void moe_zero(float* __restrict__ out, int* __restrict__ counts,
                         size_t n4) {
    size_t i = (size_t)blockIdx.x * blockDim.x + threadIdx.x;
    if (i < n4) ((float4*)out)[i] = make_float4(0.f, 0.f, 0.f, 0.f);
    if (i < NEXP) counts[i] = 0;
}

// ---------------------------------------------------------------------------
// Kernel 1: gating. One wave per token: logits, top-2, softmax, atomic append.
// ---------------------------------------------------------------------------
__global__ __launch_bounds__(256)
void moe_gate(const float* __restrict__ x,
              const float* __restrict__ gw,   // [D][E]
              const float* __restrict__ gb,   // [E]
              int* __restrict__ counts,
              int* __restrict__ bucket,       // [E][N]
              float* __restrict__ bprob) {    // [E][N]
    const int wave = threadIdx.x >> 5;
    const int lane = threadIdx.x & 31;
    const int t = blockIdx.x * 8 + wave;
    if (t >= NTOK) return;

    float acc[NEXP];
#pragma unroll
    for (int e = 0; e < NEXP; ++e) acc[e] = 0.f;

    for (int d = lane; d < DIM; d += 32) {
        float xv = x[(size_t)t * DIM + d];
        const float4* g4 = (const float4*)(gw + (size_t)d * NEXP);
        float4 g0 = g4[0], g1 = g4[1];
        acc[0] += xv * g0.x; acc[1] += xv * g0.y;
        acc[2] += xv * g0.z; acc[3] += xv * g0.w;
        acc[4] += xv * g1.x; acc[5] += xv * g1.y;
        acc[6] += xv * g1.z; acc[7] += xv * g1.w;
    }
#pragma unroll
    for (int e = 0; e < NEXP; ++e) {
#pragma unroll
        for (int off = 16; off > 0; off >>= 1)
            acc[e] += __shfl_down(acc[e], off, 32);
    }
    if (lane == 0) {
        float v[NEXP];
#pragma unroll
        for (int e = 0; e < NEXP; ++e) v[e] = acc[e] + gb[e];
        int e0 = 0;
#pragma unroll
        for (int e = 1; e < NEXP; ++e) if (v[e] > v[e0]) e0 = e;
        int e1 = (e0 == 0) ? 1 : 0;
#pragma unroll
        for (int e = 0; e < NEXP; ++e)
            if (e != e0 && v[e] > v[e1]) e1 = e;
        float x1 = __expf(v[e1] - v[e0]);
        float inv = 1.f / (1.f + x1);

        int pos0 = atomicAdd(&counts[e0], 1);
        bucket[e0 * NTOK + pos0] = t * 2 + 0;
        bprob [e0 * NTOK + pos0] = inv;
        int pos1 = atomicAdd(&counts[e1], 1);
        bucket[e1 * NTOK + pos1] = t * 2 + 1;
        bprob [e1 * NTOK + pos1] = x1 * inv;
    }
}

// ---------------------------------------------------------------------------
// Kernel 2: h = relu(gather(x) @ w1[e] + b1[e]) -> hbuf (bf16) [2N][H]
// 256 threads = 8 waves; tile 128x128, K-step 32, double-buffered LDS.
// ---------------------------------------------------------------------------
__global__ __launch_bounds__(256)
void moe_ffn1(const float* __restrict__ x,
              const float* __restrict__ w1,    // [E][D][H]
              const float* __restrict__ bias1, // [E][H]
              const int*   __restrict__ counts,
              const int*   __restrict__ bucket,
              __bf16*      __restrict__ hbuf) {
    const int e  = blockIdx.z;
    const int mt = blockIdx.y;
    const int nt = blockIdx.x;
    const int cnt = counts[e];
    if (mt * BM >= cnt) return;

    __shared__ __bf16 As[2][BM][BK + APAD];
    __shared__ __bf16 Bs[2][BN][BK + APAD];
    __shared__ __bf16 Cs[BM][BN + CPAD];
    __shared__ int sTs[BM];

    const int tid = threadIdx.x;
    if (tid < BM) {
        int pos = mt * BM + tid;
        sTs[tid] = (pos < cnt) ? bucket[e * NTOK + pos] : -1;
    }
    __syncthreads();

    // ---- per-thread gather pointers (hoisted out of K loop).
    // Invalid rows read token 0 (garbage results are never stored).
    const float* aPtr[4];
    int aM[4], aK4[4];
#pragma unroll
    for (int it = 0; it < 4; ++it) {
        int idx = tid + it * 256;          // (m, k4) over 128x8 float4s
        int m = idx >> 3, k4 = idx & 7;
        int ts = sTs[m];
        int tok = (ts < 0 ? 0 : ts) >> 1;
        aPtr[it] = x + (size_t)tok * DIM + k4 * 4;
        aM[it] = m; aK4[it] = k4;
    }
    const int kg = tid >> 5;               // 8 K-groups of 4 rows
    const int ng = tid & 31;               // 32 N-groups of 4 cols
    const float* bPtr = w1 + (size_t)e * DIM * HID + (size_t)(kg * 4) * HID
                        + nt * BN + ng * 4;

    v4f aReg[4], bReg[4];
    auto gload = [&](int kt) {
#pragma unroll
        for (int it = 0; it < 4; ++it)
            aReg[it] = *(const v4f*)(aPtr[it] + kt);
#pragma unroll
        for (int r = 0; r < 4; ++r)
            bReg[r] = *(const v4f*)(bPtr + (size_t)(kt + r) * HID);
    };
    auto stage = [&](int b) {
#pragma unroll
        for (int it = 0; it < 4; ++it)
            *(v4bf*)&As[b][aM[it]][aK4[it] * 4] = cvt4(aReg[it]);
#pragma unroll
        for (int c = 0; c < 4; ++c) {      // 4x4 register transpose
            v4f col = { bReg[0][c], bReg[1][c], bReg[2][c], bReg[3][c] };
            *(v4bf*)&Bs[b][ng * 4 + c][kg * 4] = cvt4(col);
        }
    };

    const int lane = tid & 31;
    const int wid  = tid >> 5;
    const int wm = wid & 3;                // 4 wave rows * 32
    const int wn = wid >> 2;               // 2 wave cols * 64
    const int fr = lane & 15;
    const int kh = lane >> 4;

    v8f acc[2][4];
#pragma unroll
    for (int i = 0; i < 2; ++i)
#pragma unroll
        for (int j = 0; j < 4; ++j) acc[i][j] = (v8f)0.f;

    gload(0); stage(0);
    __syncthreads();

    int cur = 0;
    for (int kt = 0; kt < DIM; kt += BK) {
        const bool more = (kt + BK) < DIM;
        if (more) {
            gload(kt + BK);
            if (kt + 2 * BK < DIM)
                __builtin_prefetch(bPtr + (size_t)(kt + 2 * BK) * HID, 0, 1);
        }

        v16bf af[2], bf[4];
#pragma unroll
        for (int i = 0; i < 2; ++i) {
            const __bf16* s = &As[cur][wm * 32 + i * 16 + fr][kh * 8];
            af[i] = comb16(*(const v8bf*)s, *(const v8bf*)(s + 16));
        }
#pragma unroll
        for (int j = 0; j < 4; ++j) {
            const __bf16* s = &Bs[cur][wn * 64 + j * 16 + fr][kh * 8];
            bf[j] = comb16(*(const v8bf*)s, *(const v8bf*)(s + 16));
        }
#pragma unroll
        for (int i = 0; i < 2; ++i)
#pragma unroll
            for (int j = 0; j < 4; ++j)
                acc[i][j] = __builtin_amdgcn_wmma_f32_16x16x32_bf16(
                    false, af[i], false, bf[j], (short)0, acc[i][j],
                    false, false);

        if (more) {
            stage(cur ^ 1);
            __syncthreads();
            cur ^= 1;
        }
    }

    // ---- epilogue: bias + relu -> Cs, then coalesced bf16 copy to hbuf
#pragma unroll
    for (int j = 0; j < 4; ++j) {
        int col = wn * 64 + j * 16 + fr;
        float b = bias1[(size_t)e * HID + nt * BN + col];
#pragma unroll
        for (int i = 0; i < 2; ++i)
#pragma unroll
            for (int r = 0; r < 8; ++r) {
                int m = wm * 32 + i * 16 + r + 8 * kh;
                float v = acc[i][j][r] + b;
                Cs[m][col] = (__bf16)(v > 0.f ? v : 0.f);
            }
    }
    __syncthreads();
    {
        int m = tid >> 1, half = tid & 1;
        int ts = sTs[m];
        if (ts >= 0) {
            __bf16* dst = hbuf + (size_t)ts * HID + nt * BN + half * 64;
            const __bf16* src = &Cs[m][half * 64];
#pragma unroll
            for (int q = 0; q < 8; ++q)
                *(uint4*)(dst + q * 8) = *(const uint4*)(src + q * 8);
        }
    }
}

// ---------------------------------------------------------------------------
// Kernel 3: y = gather(h) @ w2[e] + b2[e]; out[token] += prob * y (atomic).
// A operand is bf16 in memory -> async copy straight to LDS when available.
// ---------------------------------------------------------------------------
__global__ __launch_bounds__(256)
void moe_ffn2(const __bf16* __restrict__ hbuf,  // [2N][H]
              const float*  __restrict__ w2,    // [E][H][O]
              const float*  __restrict__ bias2, // [E][O]
              const int*    __restrict__ counts,
              const int*    __restrict__ bucket,
              const float*  __restrict__ bprob,
              float*        __restrict__ out) {
    const int e  = blockIdx.z;
    const int mt = blockIdx.y;
    const int nt = blockIdx.x;
    const int cnt = counts[e];
    if (mt * BM >= cnt) return;

    __shared__ __bf16 As[2][BM][BK + APAD];
    __shared__ __bf16 Bs[2][BN][BK + APAD];
    __shared__ int   sTs[BM];
    __shared__ float sPr[BM];

    const int tid = threadIdx.x;
    if (tid < BM) {
        int pos = mt * BM + tid;
        bool ok = pos < cnt;
        sTs[tid] = ok ? bucket[e * NTOK + pos] : -1;
        sPr[tid] = ok ? bprob [e * NTOK + pos] : 0.f;
    }
    __syncthreads();

    // ---- hoisted gather pointers: 2 chunks of 16B per thread (128x32 bf16)
    const __bf16* aSrc[2];
    int aM[2], aC[2];
#pragma unroll
    for (int it = 0; it < 2; ++it) {
        int idx = tid + it * 256;          // (m, c) over 128x4 uint4s
        int m = idx >> 2, c = idx & 3;
        int ts = sTs[m];
        int tsc = ts < 0 ? 0 : ts;
        aSrc[it] = hbuf + (size_t)tsc * HID + c * 8;
        aM[it] = m; aC[it] = c;
    }
    const int kg = tid >> 5;
    const int ng = tid & 31;
    const float* bPtr = w2 + (size_t)e * HID * OUTD + (size_t)(kg * 4) * OUTD
                        + nt * BN + ng * 4;

#if !HAS_ASYNC_LDS
    uint4 aReg[2];
#endif
    v4f bReg[4];

    auto stageA_issue = [&](int b, int kt) {
#if HAS_ASYNC_LDS
#pragma unroll
        for (int it = 0; it < 2; ++it)
            __builtin_amdgcn_global_load_async_to_lds_b128(
                (GAS1 v4i*)(aSrc[it] + kt),
                (GAS3 v4i*)&As[b][aM[it]][aC[it] * 8],
                0, 0);
#else
#pragma unroll
        for (int it = 0; it < 2; ++it)
            aReg[it] = *(const uint4*)(aSrc[it] + kt);
        (void)b;
#endif
    };
    auto stageA_finish = [&](int b) {
#if HAS_ASYNC_LDS
        wait_async0();
        (void)b;
#else
#pragma unroll
        for (int it = 0; it < 2; ++it)
            *(uint4*)&As[b][aM[it]][aC[it] * 8] = aReg[it];
#endif
    };
    auto loadB = [&](int kt) {
#pragma unroll
        for (int r = 0; r < 4; ++r)
            bReg[r] = *(const v4f*)(bPtr + (size_t)(kt + r) * OUTD);
    };
    auto stageB = [&](int b) {
#pragma unroll
        for (int c = 0; c < 4; ++c) {
            v4f col = { bReg[0][c], bReg[1][c], bReg[2][c], bReg[3][c] };
            *(v4bf*)&Bs[b][ng * 4 + c][kg * 4] = cvt4(col);
        }
    };

    const int lane = tid & 31;
    const int wid  = tid >> 5;
    const int wm = wid & 3;
    const int wn = wid >> 2;
    const int fr = lane & 15;
    const int kh = lane >> 4;

    v8f acc[2][4];
#pragma unroll
    for (int i = 0; i < 2; ++i)
#pragma unroll
        for (int j = 0; j < 4; ++j) acc[i][j] = (v8f)0.f;

    stageA_issue(0, 0); loadB(0); stageB(0); stageA_finish(0);
    __syncthreads();

    int cur = 0;
    for (int kt = 0; kt < HID; kt += BK) {
        const bool more = (kt + BK) < HID;
        if (more) {
            stageA_issue(cur ^ 1, kt + BK);
            loadB(kt + BK);
            if (kt + 2 * BK < HID)
                __builtin_prefetch(bPtr + (size_t)(kt + 2 * BK) * OUTD, 0, 1);
        }

        v16bf af[2], bf[4];
#pragma unroll
        for (int i = 0; i < 2; ++i) {
            const __bf16* s = &As[cur][wm * 32 + i * 16 + fr][kh * 8];
            af[i] = comb16(*(const v8bf*)s, *(const v8bf*)(s + 16));
        }
#pragma unroll
        for (int j = 0; j < 4; ++j) {
            const __bf16* s = &Bs[cur][wn * 64 + j * 16 + fr][kh * 8];
            bf[j] = comb16(*(const v8bf*)s, *(const v8bf*)(s + 16));
        }
#pragma unroll
        for (int i = 0; i < 2; ++i)
#pragma unroll
            for (int j = 0; j < 4; ++j)
                acc[i][j] = __builtin_amdgcn_wmma_f32_16x16x32_bf16(
                    false, af[i], false, bf[j], (short)0, acc[i][j],
                    false, false);

        if (more) {
            stageB(cur ^ 1);
            stageA_finish(cur ^ 1);
            __syncthreads();
            cur ^= 1;
        }
    }

    // ---- epilogue: (acc + b2) * prob, atomic scatter-add into out
#pragma unroll
    for (int j = 0; j < 4; ++j) {
        int col = wn * 64 + j * 16 + fr;
        float b = bias2[(size_t)e * OUTD + nt * BN + col];
#pragma unroll
        for (int i = 0; i < 2; ++i)
#pragma unroll
            for (int r = 0; r < 8; ++r) {
                int m = wm * 32 + i * 16 + r + 8 * kh;
                int ts = sTs[m];
                if (ts >= 0) {
                    float v = (acc[i][j][r] + b) * sPr[m];
                    atomicAdd(out + (size_t)(ts >> 1) * OUTD + nt * BN + col, v);
                }
            }
    }
}

// ---------------------------------------------------------------------------
// Launch
// ---------------------------------------------------------------------------
extern "C" void kernel_launch(void* const* d_in, const int* in_sizes, int n_in,
                              void* d_out, int out_size, void* d_ws,
                              size_t ws_size, hipStream_t stream) {
    const float* x  = (const float*)d_in[0];
    const float* gw = (const float*)d_in[1];
    const float* gb = (const float*)d_in[2];
    const float* w1 = (const float*)d_in[3];
    const float* b1 = (const float*)d_in[4];
    const float* w2 = (const float*)d_in[5];
    const float* b2 = (const float*)d_in[6];
    // d_in[7] == k (2) : hardcoded
    float* out = (float*)d_out;

    char* ws = (char*)d_ws;
    int*    counts = (int*)ws;                                   // 32 B
    int*    bucket = (int*)(ws + 256);                           // 256 KB
    float*  bprob  = (float*)(ws + 256 + NEXP * NTOK * 4);       // 256 KB
    __bf16* hbuf   = (__bf16*)(ws + (1 << 20));                  // 64 MB

    size_t n4 = (size_t)NTOK * OUTD / 4;
    moe_zero<<<dim3((unsigned)((n4 + 255) / 256)), 256, 0, stream>>>(out, counts, n4);
    moe_gate<<<NTOK / 8, 256, 0, stream>>>(x, gw, gb, counts, bucket, bprob);

    dim3 g1(HID / BN, NTOK / BM, NEXP);
    moe_ffn1<<<g1, 256, 0, stream>>>(x, w1, b1, counts, bucket, hbuf);

    dim3 g2(OUTD / BN, NTOK / BM, NEXP);
    moe_ffn2<<<g2, 256, 0, stream>>>(hbuf, w2, b2, counts, bucket, bprob, out);
}
